// Encoder_7928509629181
// MI455X (gfx1250) — matrix operand order repaired and checked by reading, not compiled, not run on hardware
//
#include <hip/hip_runtime.h>

// Problem dimensions (compile-time constants from the reference)
#define B_   256
#define T_   512
#define E_   256
#define H_   512
#define I_   512   // 2*E
#define G4H  2048  // 4*H

typedef __attribute__((ext_vector_type(16))) __bf16       v16bf;
typedef __attribute__((ext_vector_type(8)))  float        v8f;
typedef __attribute__((ext_vector_type(8)))  unsigned int v8u;

// ---------- helpers ----------

__device__ __forceinline__ unsigned short f32_to_bf16(float f) {
  unsigned int u = __builtin_bit_cast(unsigned int, f);
  u += 0x7fffu + ((u >> 16) & 1u);           // round-to-nearest-even
  return (unsigned short)(u >> 16);
}

// monotonic order-preserving float<->uint mapping (for packed atomic max)
__device__ __forceinline__ unsigned int ordered_f32(float f) {
  unsigned int u = __builtin_bit_cast(unsigned int, f);
  return (u & 0x80000000u) ? ~u : (u | 0x80000000u);
}
__device__ __forceinline__ float unordered_f32(unsigned int o) {
  unsigned int u = (o & 0x80000000u) ? (o & 0x7fffffffu) : ~o;
  return __builtin_bit_cast(float, u);
}

// A-matrix 16x32 bf16 fragment: lanes 0-15 -> K 0..15 ;
// lanes 16-31 -> K 16..31 ; M = lane & 15. Row-major source [M][ldk].
__device__ __forceinline__ v16bf load_a_frag(const unsigned short* __restrict__ base,
                                             int ldk, int m0, int k0, int lane) {
  const int half = lane >> 4, mr = lane & 15;
  const unsigned short* row = base + (size_t)(m0 + mr) * ldk + k0 + half * 16;
  v8u t;
#pragma unroll
  for (int v = 0; v < 8; ++v) {
    const int k = v * 2;        // pair K offset
    t[v] = *(const unsigned int*)(row + k);                // 2 bf16 per dword
  }
  return __builtin_bit_cast(v16bf, t);
}

// B-matrix 32x16 bf16 fragment: lane&15 = N column, lane>>4 selects K 0-15 / 16-31,
// VGPR v holds K = 2v, 2v+1.  B[k][n] = W[nbase+n][k] (row-major W, [N][ldk]).
__device__ __forceinline__ v16bf load_b_frag(const unsigned short* __restrict__ w,
                                             int ldk, int nbase, int k0, int lane) {
  const int n = lane & 15, khalf = lane >> 4;
  const unsigned short* row = w + (size_t)(nbase + n) * ldk + k0 + khalf * 16;
  v8u t;
#pragma unroll
  for (int p = 0; p < 8; ++p)
    t[p] = *(const unsigned int*)(row + p * 2);
  return __builtin_bit_cast(v16bf, t);
}

__device__ __forceinline__ v8f wmma_bf16(v16bf a, v16bf b, v8f c) {
  return __builtin_amdgcn_wmma_f32_16x16x32_bf16(false, a, false, b,
                                                 (short)0, c, false, false);
}

__device__ __forceinline__ float sigmoidf_(float x) {
  return 1.f / (1.f + __expf(-x));
}

// ---------- prep kernels ----------

__global__ void f2bf_kernel(const float* __restrict__ src,
                            unsigned short* __restrict__ dst, int n) {
  for (int i = blockIdx.x * blockDim.x + threadIdx.x; i < n;
       i += gridDim.x * blockDim.x)
    dst[i] = f32_to_bf16(src[i]);
}

__global__ void bias_kernel(const float* __restrict__ a, const float* __restrict__ b,
                            float* __restrict__ o, int n) {
  for (int i = blockIdx.x * blockDim.x + threadIdx.x; i < n;
       i += gridDim.x * blockDim.x)
    o[i] = a[i] + b[i];
}

// xcat[t][b][c] (bf16) from feat/age [b][t][e]
__global__ void pack_x_kernel(const float* __restrict__ feat,
                              const float* __restrict__ age,
                              unsigned short* __restrict__ xcat) {
  const long long N = (long long)T_ * B_ * I_;
  const long long stride = (long long)gridDim.x * blockDim.x;
  for (long long i = blockIdx.x * (long long)blockDim.x + threadIdx.x; i < N;
       i += stride) {
    const int c = (int)(i % I_);
    const long long r = i / I_;
    const int b = (int)(r % B_);
    const int t = (int)(r / B_);
    const float v = (c < E_) ? feat[((size_t)b * T_ + t) * E_ + c]
                             : age[((size_t)b * T_ + t) * E_ + (c - E_)];
    xcat[i] = f32_to_bf16(v);
  }
}

__global__ void init_state_kernel(unsigned short* __restrict__ hbf0,
                                  unsigned short* __restrict__ hbf1,
                                  float* __restrict__ h32, float* __restrict__ c32,
                                  unsigned long long* __restrict__ pool) {
  const int n = B_ * H_;  // == B_*T_
  for (int i = blockIdx.x * blockDim.x + threadIdx.x; i < n;
       i += gridDim.x * blockDim.x) {
    hbf0[i] = 0; hbf1[i] = 0; h32[i] = 0.f; c32[i] = 0.f; pool[i] = 0ull;
  }
}

// ---------- x_proj = xcat @ W_ih^T + (b_ih + b_hh) ----------
// M = T*B = 131072, N = 2048, K = 512.  8 waves/block, each wave: 16x64 tile.
__global__ void xproj_gemm_kernel(const unsigned short* __restrict__ x_bf,
                                  const unsigned short* __restrict__ wih_bf,
                                  const float* __restrict__ bias,
                                  float* __restrict__ y) {
  const int wave = threadIdx.x >> 5, lane = threadIdx.x & 31;
  const int m0 = blockIdx.x * 128 + wave * 16;
  const int n0 = blockIdx.y * 64;
  v8f acc[4] = {};
  for (int k = 0; k < I_; k += 32) {
    const v16bf a = load_a_frag(x_bf, I_, m0, k, lane);
#pragma unroll
    for (int j = 0; j < 4; ++j) {
      const v16bf b = load_b_frag(wih_bf, I_, n0 + j * 16, k, lane);
      acc[j] = wmma_bf16(a, b, acc[j]);
    }
  }
  const int nl = lane & 15, mh = (lane >> 4) * 8;   // C/D layout
#pragma unroll
  for (int j = 0; j < 4; ++j) {
    const int n = n0 + j * 16 + nl;
    const float bv = bias[n];
#pragma unroll
    for (int r = 0; r < 8; ++r)
      y[(size_t)(m0 + mh + r) * G4H + n] = acc[j][r] + bv;
  }
}

// ---------- one LSTM timestep, fused gates + nonlinearity + pooling ----------
// grid (B/16, H/64), 256 threads = 8 waves; wave w -> gate (w>>1), col half (w&1)*32
__global__ void lstm_step_kernel(const unsigned short* __restrict__ hprev_bf,
                                 const unsigned short* __restrict__ whh_bf,
                                 const float* __restrict__ xproj_t, // [B][4H]
                                 const int* __restrict__ lengths, int t,
                                 unsigned short* __restrict__ hnext_bf,
                                 float* __restrict__ h32, float* __restrict__ c32,
                                 unsigned long long* __restrict__ pool) {
  __shared__ float gsm[4][16][64];                 // 16 KB gate accumulators
  __shared__ unsigned long long red[16][16];       // pooling reduction
  const int tid = threadIdx.x;
  const int wave = tid >> 5, lane = tid & 31;
  const int m0 = blockIdx.x * 16;                  // batch tile
  const int n0 = blockIdx.y * 64;                  // hidden-col tile
  const int g = wave >> 1;                         // gate 0..3 (i,f,g,o)
  const int nsub = (wave & 1) * 32;

  v8f acc0 = {}, acc1 = {};
  for (int k = 0; k < H_; k += 32) {
    const v16bf a  = load_a_frag(hprev_bf, H_, m0, k, lane);
    const v16bf b0 = load_b_frag(whh_bf, H_, g * H_ + n0 + nsub,      k, lane);
    const v16bf b1 = load_b_frag(whh_bf, H_, g * H_ + n0 + nsub + 16, k, lane);
    acc0 = wmma_bf16(a, b0, acc0);
    acc1 = wmma_bf16(a, b1, acc1);
  }
  {
    const int nl = lane & 15, mh = (lane >> 4) * 8;
#pragma unroll
    for (int r = 0; r < 8; ++r) {
      gsm[g][mh + r][nsub + nl]      = acc0[r];
      gsm[g][mh + r][nsub + 16 + nl] = acc1[r];
    }
  }
  __syncthreads();

  // elementwise: thread -> row m = tid>>4, 4 consecutive cols
  const int m  = tid >> 4;
  const int nq = (tid & 15) << 2;
  const int bm = m0 + m;
  const int len = lengths[bm];
  const bool valid = (t < len);
  unsigned long long best = 0ull;
  const float* xp = xproj_t + (size_t)bm * G4H;
#pragma unroll
  for (int q = 0; q < 4; ++q) {
    const int n = nq + q;
    const int nc = n0 + n;
    const size_t off = (size_t)bm * H_ + nc;
    const float gi = gsm[0][m][n] + xp[nc];
    const float gf = gsm[1][m][n] + xp[H_ + nc];
    const float gg = gsm[2][m][n] + xp[2 * H_ + nc];
    const float go = gsm[3][m][n] + xp[3 * H_ + nc];
    const float i_ = sigmoidf_(gi);
    const float f_ = sigmoidf_(gf);
    const float g_ = tanhf(gg);
    const float o_ = sigmoidf_(go);
    const float cp = c32[off];
    const float cn = f_ * cp + i_ * g_;
    const float hn = o_ * tanhf(cn);
    const float hsel = valid ? hn : h32[off];
    const float csel = valid ? cn : cp;
    h32[off] = hsel;
    c32[off] = csel;
    hnext_bf[off] = f32_to_bf16(hsel);
    // reference reshape quirk (T==H): pool over hidden index nc < len,
    // value is masked out_h = valid ? hn : 0. argmax-first via ~nc in low bits.
    if (nc < len) {
      const float outv = valid ? hn : 0.f;
      const unsigned long long p =
          ((unsigned long long)ordered_f32(outv) << 32) |
          (unsigned long long)(unsigned)(~(unsigned)nc);
      if (p > best) best = p;
    }
  }
  red[m][tid & 15] = best;
  __syncthreads();
  if ((tid & 15) == 0) {
    unsigned long long v = red[m][0];
#pragma unroll
    for (int q = 1; q < 16; ++q) {
      const unsigned long long w2 = red[m][q];
      if (w2 > v) v = w2;
    }
    if (v) atomicMax(&pool[(size_t)bm * T_ + t], v);
  }
}

// ---------- finalize: code_pool, code_indices, h_n, c_n ----------
__global__ void finalize_kernel(const unsigned long long* __restrict__ pool,
                                const float* __restrict__ h32,
                                const float* __restrict__ c32,
                                float* __restrict__ out) {
  const int n = B_ * H_;
  for (int i = blockIdx.x * blockDim.x + threadIdx.x; i < n;
       i += gridDim.x * blockDim.x) {
    const unsigned long long p = pool[i];
    float val = 0.f, idx = 0.f;
    if (p) {
      val = unordered_f32((unsigned int)(p >> 32));
      idx = (float)(~(unsigned int)p);
    }
    out[i]         = val;        // code_pool
    out[n + i]     = idx;        // code_indices (as float)
    out[2 * n + i] = h32[i];     // h_n
    out[3 * n + i] = c32[i];     // c_n
  }
}

// ---------- host ----------

extern "C" void kernel_launch(void* const* d_in, const int* in_sizes, int n_in,
                              void* d_out, int out_size, void* d_ws, size_t ws_size,
                              hipStream_t stream) {
  (void)in_sizes; (void)n_in; (void)out_size; (void)ws_size;
  const float* feat    = (const float*)d_in[0];
  const float* age     = (const float*)d_in[1];
  const int*   lengths = (const int*)d_in[2];
  const float* W_ih    = (const float*)d_in[3];
  const float* W_hh    = (const float*)d_in[4];
  const float* b_ih    = (const float*)d_in[5];
  const float* b_hh    = (const float*)d_in[6];
  float* out = (float*)d_out;

  char* ws = (char*)d_ws;
  size_t off = 0;
  auto carve = [&](size_t bytes) {
    char* p = ws + off;
    off += (bytes + 255) & ~(size_t)255;
    return p;
  };
  unsigned short* wih_bf = (unsigned short*)carve((size_t)G4H * I_ * 2);
  unsigned short* whh_bf = (unsigned short*)carve((size_t)G4H * H_ * 2);
  float*          bias   = (float*)carve((size_t)G4H * 4);
  unsigned short* xcat   = (unsigned short*)carve((size_t)T_ * B_ * I_ * 2);
  float*          xproj  = (float*)carve((size_t)T_ * B_ * G4H * 4);
  unsigned short* hbf    = (unsigned short*)carve((size_t)2 * B_ * H_ * 2);
  float*          h32    = (float*)carve((size_t)B_ * H_ * 4);
  float*          c32    = (float*)carve((size_t)B_ * H_ * 4);
  unsigned long long* pool = (unsigned long long*)carve((size_t)B_ * T_ * 8);

  f2bf_kernel<<<1024, 256, 0, stream>>>(W_ih, wih_bf, G4H * I_);
  f2bf_kernel<<<1024, 256, 0, stream>>>(W_hh, whh_bf, G4H * H_);
  bias_kernel<<<8, 256, 0, stream>>>(b_ih, b_hh, bias, G4H);
  pack_x_kernel<<<4096, 256, 0, stream>>>(feat, age, xcat);
  init_state_kernel<<<512, 256, 0, stream>>>(hbf, hbf + (size_t)B_ * H_, h32, c32, pool);

  xproj_gemm_kernel<<<dim3((T_ * B_) / 128, G4H / 64), 256, 0, stream>>>(
      xcat, wih_bf, bias, xproj);

  for (int t = 0; t < T_; ++t) {
    const unsigned short* hp = hbf + (size_t)(t & 1) * B_ * H_;
    unsigned short*       hn = hbf + (size_t)((t + 1) & 1) * B_ * H_;
    lstm_step_kernel<<<dim3(B_ / 16, H_ / 64), 256, 0, stream>>>(
        hp, whh_bf, xproj + (size_t)t * B_ * G4H, lengths, t, hn, h32, c32, pool);
  }

  finalize_kernel<<<512, 256, 0, stream>>>(pool, h32, c32, out);
}
